// DeepFuzzyCMeanModel_18494129176961
// MI455X (gfx1250) — matrix operand order repaired and checked by reading, not compiled, not run on hardware
//
#include <hip/hip_runtime.h>

// ---------------------------------------------------------------------------
// Fuzzy c-means loss, reformulated so the N-reduction happens inside WMMA:
//   loss = G * sum_{k,d} ( W2[k,d] + c[k]*v[k,d]^2 - 2*W[k,d]*v[k,d] )
//   W  = (u^2)^T @ x       [K,D]
//   W2 = (u^2)^T @ (x*x)   [K,D]
//   c  = sum_n u^2[n,:]    [K]
// x,u streamed exactly once (201 MB -> ~8.6us HBM floor @ 23.3 TB/s).
// GEMMs: v_wmma_f32_16x16x32_bf16 (bf16 in, f32 accumulate).
// Staging: double-buffered GLOBAL_LOAD_ASYNC_TO_LDS_B128 (ASYNCcnt-tracked),
// overlapping next-chunk HBM->LDS DMA with current-chunk WMMA.
// ---------------------------------------------------------------------------

typedef __attribute__((ext_vector_type(16))) __bf16 v16bf;
typedef __attribute__((ext_vector_type(8)))  float  v8f;
typedef int v4i __attribute__((vector_size(16)));   // matches builtin param

#define GAMMA_F     1e-6f
#define D_DIM       128
#define K_DIM       64
#define CHUNK_ROWS  32
#define THREADS     256
#define GEMM_BLOCKS 512

// workspace layout (floats): W[K*D] | W2[K*D] | C[K]
#define WS_W        0
#define WS_W2       (K_DIM * D_DIM)
#define WS_C        (2 * K_DIM * D_DIM)
#define WS_FLOATS   (2 * K_DIM * D_DIM + K_DIM)

#if defined(__HIP_DEVICE_COMPILE__) && \
    __has_builtin(__builtin_amdgcn_global_load_async_to_lds_b128)
#define HAVE_ASYNC_LDS 1
#else
#define HAVE_ASYNC_LDS 0
#endif

__global__ __launch_bounds__(THREADS)
void fcm_zero_ws(float* __restrict__ ws) {
    int i = blockIdx.x * THREADS + threadIdx.x;
    if (i < WS_FLOATS) ws[i] = 0.0f;
}

// Issue one chunk's global->LDS staging (async if available, else direct).
__device__ __forceinline__ void stage_chunk(const float* __restrict__ x,
                                            const float* __restrict__ u,
                                            size_t n0,
                                            float* __restrict__ bx,
                                            float* __restrict__ bu,
                                            int tid) {
    const float4* xg = (const float4*)(x + n0 * D_DIM); // 1024 float4
    const float4* ug = (const float4*)(u + n0 * K_DIM); //  512 float4
    float4* xs = (float4*)bx;
    float4* us = (float4*)bu;
#if HAVE_ASYNC_LDS
    #pragma unroll
    for (int i = 0; i < 4; ++i) {
        const int idx = tid + i * THREADS;
        __builtin_amdgcn_global_load_async_to_lds_b128(
            (__attribute__((address_space(1))) v4i*)(xg + idx),
            (__attribute__((address_space(3))) v4i*)(xs + idx), 0, 0);
    }
    #pragma unroll
    for (int i = 0; i < 2; ++i) {
        const int idx = tid + i * THREADS;
        __builtin_amdgcn_global_load_async_to_lds_b128(
            (__attribute__((address_space(1))) v4i*)(ug + idx),
            (__attribute__((address_space(3))) v4i*)(us + idx), 0, 0);
    }
#else
    #pragma unroll
    for (int i = 0; i < 4; ++i) xs[tid + i * THREADS] = xg[tid + i * THREADS];
    #pragma unroll
    for (int i = 0; i < 2; ++i) us[tid + i * THREADS] = ug[tid + i * THREADS];
#endif
}

__device__ __forceinline__ void wait_async_done() {
#if HAVE_ASYNC_LDS
  #if __has_builtin(__builtin_amdgcn_s_wait_asynccnt)
    __builtin_amdgcn_s_wait_asynccnt(0);
  #else
    asm volatile("s_wait_asynccnt 0x0" ::: "memory");
  #endif
#endif
}

__global__ __launch_bounds__(THREADS)
void fcm_gemm(const float* __restrict__ x,   // [N,128]
              const float* __restrict__ u,   // [N,64]
              float* __restrict__ ws,
              int nchunks) {
    __shared__ float lds_x[2][CHUNK_ROWS * D_DIM]; // 2 x 16 KB
    __shared__ float lds_u[2][CHUNK_ROWS * K_DIM]; // 2 x  8 KB

    const int tid  = threadIdx.x;
    const int lane = tid & 31;
    const int w    = tid >> 5;      // wave 0..7
    const int hb   = lane >> 4;     // half-wave select (0/1)
    const int l15  = lane & 15;
    const int kt   = w & 3;         // k-tile 0..3  (k = 16*kt + ..)
    const int dt0  = (w >> 2) * 4;  // first of 4 owned d-tiles

    v8f acc[4]  = {};               // W  tiles, f32 accum across all chunks
    v8f acc2[4] = {};               // W2 tiles
    float c_acc = 0.0f;             // column sums of u^2 (waves 0..3 only)

    int parity = 0;
    if (blockIdx.x < nchunks)
        stage_chunk(x, u, (size_t)blockIdx.x * CHUNK_ROWS,
                    lds_x[0], lds_u[0], tid);

    for (int c = blockIdx.x; c < nchunks; c += gridDim.x) {
        wait_async_done();   // this wave's chunk-c transfers have landed
        __syncthreads();     // whole block's transfers landed; prior compute
                             // on the opposite buffer is complete everywhere

        const int cnext = c + gridDim.x;
        if (cnext < nchunks)
            stage_chunk(x, u, (size_t)cnext * CHUNK_ROWS,
                        lds_x[parity ^ 1], lds_u[parity ^ 1], tid);

        const float* __restrict__ bx = lds_x[parity];
        const float* __restrict__ bu = lds_u[parity];

        // ---- A fragment: 16x32 bf16 tile of (u^2)^T, M=k row, K=n ----
        // ISA layout: lanes 0-15 (hb=0) hold K in {e | e<8} U {16+(e-8)};
        // lanes 16-31 (hb=1) shift both groups by +8.
        v16bf abf;
        float csum = 0.0f;
        #pragma unroll
        for (int e = 0; e < 16; ++e) {
            const int n = (e < 8) ? (8 * hb + e) : (8 + 8 * hb + e);
            const float uv = bu[n * K_DIM + kt * 16 + l15];
            const float a  = uv * uv;
            csum  += a;
            abf[e] = (__bf16)a;
        }
        if (w < 4) c_acc += csum;   // each (k, n-half) counted exactly once

        // ---- B fragments: 32x16 bf16 tiles of x (and x*x), K=n, N=d ----
        // ISA layout: lane holds column N=l15, K in {8*hb+e | e<8} U {16+8*hb+(e-8)}.
        #pragma unroll
        for (int j = 0; j < 4; ++j) {
            const int dcol = (dt0 + j) * 16 + l15;
            v16bf bbf, b2bf;
            #pragma unroll
            for (int e = 0; e < 16; ++e) {
                const float xv = bx[((e < 8) ? (8 * hb + e) : (8 + 8 * hb + e)) * D_DIM + dcol];
                bbf[e]  = (__bf16)xv;
                b2bf[e] = (__bf16)(xv * xv);
            }
            acc[j]  = __builtin_amdgcn_wmma_f32_16x16x32_bf16(
                          false, abf, false, bbf,  (short)0, acc[j],  false, false);
            acc2[j] = __builtin_amdgcn_wmma_f32_16x16x32_bf16(
                          false, abf, false, b2bf, (short)0, acc2[j], false, false);
        }
        parity ^= 1;
        // no trailing barrier: the barrier at the top of the next iteration
        // protects the buffer being staged into.
    }

    // ---- write partial results (C/D layout: VGPR r, lane -> M=r+8*hb, N=l15) ----
    float* Wg  = ws + WS_W;
    float* W2g = ws + WS_W2;
    float* Cg  = ws + WS_C;
    #pragma unroll
    for (int j = 0; j < 4; ++j) {
        const int d = (dt0 + j) * 16 + l15;
        #pragma unroll
        for (int r = 0; r < 8; ++r) {
            const int k = kt * 16 + 8 * hb + r;
            atomicAdd(&Wg [k * D_DIM + d], acc[j][r]);
            atomicAdd(&W2g[k * D_DIM + d], acc2[j][r]);
        }
    }
    if (w < 4) atomicAdd(&Cg[kt * 16 + l15], c_acc);
}

__global__ __launch_bounds__(THREADS)
void fcm_finalize(const float* __restrict__ ws,
                  const float* __restrict__ v,   // [K,D] f32 (exact path)
                  float* __restrict__ out) {
    __shared__ float red[THREADS];
    const float* Wg  = ws + WS_W;
    const float* W2g = ws + WS_W2;
    const float* Cg  = ws + WS_C;

    float p = 0.0f;
    for (int i = threadIdx.x; i < K_DIM * D_DIM; i += THREADS) {
        const int   k  = i >> 7;            // i / D_DIM
        const float vv = v[i];
        p += W2g[i] + Cg[k] * vv * vv - 2.0f * Wg[i] * vv;
    }
    red[threadIdx.x] = p;
    __syncthreads();
    for (int s = THREADS / 2; s > 0; s >>= 1) {
        if (threadIdx.x < s) red[threadIdx.x] += red[threadIdx.x + s];
        __syncthreads();
    }
    if (threadIdx.x == 0) out[0] = GAMMA_F * red[0];
}

extern "C" void kernel_launch(void* const* d_in, const int* in_sizes, int n_in,
                              void* d_out, int out_size, void* d_ws, size_t ws_size,
                              hipStream_t stream) {
    const float* x = (const float*)d_in[0];   // [N,128]
    const float* u = (const float*)d_in[1];   // [N,64]
    const float* v = (const float*)d_in[2];   // [64,128]
    float* out = (float*)d_out;
    float* ws  = (float*)d_ws;                // needs WS_FLOATS*4 = 65,792 B

    const int N       = in_sizes[0] / D_DIM;
    const int nchunks = N / CHUNK_ROWS;

    fcm_zero_ws<<<(WS_FLOATS + THREADS - 1) / THREADS, THREADS, 0, stream>>>(ws);
    fcm_gemm<<<GEMM_BLOCKS, THREADS, 0, stream>>>(x, u, ws, nchunks);
    fcm_finalize<<<1, THREADS, 0, stream>>>(ws, v, out);
}
